// GNN_Layer_24524263260519
// MI455X (gfx1250) — compile-verified, matrix operand order checked by reading, not run on hardware
//
#include <hip/hip_runtime.h>
#include <stdint.h>

typedef __attribute__((ext_vector_type(16))) _Float16 v16h;
typedef __attribute__((ext_vector_type(8)))  float    v8f;
typedef __attribute__((ext_vector_type(4)))  float    v4f;

#define DFEAT 64

// ---------------------------------------------------------------------------
// Zero the h_neigh accumulator (float4 vectorized).
// ---------------------------------------------------------------------------
__global__ void zero_kernel(float4* __restrict__ p, int n4) {
  int i = blockIdx.x * blockDim.x + threadIdx.x;
  if (i < n4) p[i] = make_float4(0.f, 0.f, 0.f, 0.f);
}

// ---------------------------------------------------------------------------
// Pack W (shape [64 out, 128 in], row-major f32) into f16 WMMA B-fragments.
// Fragment addressing: frag[((c*4 + t)*32 + lane)*8 + v] (dword = 2 halves)
//   c = K-chunk (0..3, 32 K each), t = N-tile (0..3, 16 outputs each)
//   lane: column n = lane&15; K-pair index per ISA 16-bit layout:
//   kp = (v<4 ? v : v+4) + (lane>=16 ? 4 : 0); K = c*32 + 2*kp (+0/+1)
//   B[k][n] = W[t*16+n][k]
// ---------------------------------------------------------------------------
__global__ void pack_weights(const float* __restrict__ Wm, const float* __restrict__ Wa,
                             uint32_t* __restrict__ fm, uint32_t* __restrict__ fa) {
  const float*  W = (blockIdx.x == 0) ? Wm : Wa;
  uint32_t*     F = (blockIdx.x == 0) ? fm : fa;
  for (int idx = threadIdx.x; idx < 4096; idx += blockDim.x) {
    int v    = idx & 7;
    int lane = (idx >> 3) & 31;
    int t    = (idx >> 8) & 3;
    int c    = idx >> 10;
    int n    = lane & 15;
    int kp   = ((v < 4) ? v : v + 4) + ((lane >= 16) ? 4 : 0);
    int k    = c * 32 + 2 * kp;
    int o    = t * 16 + n;
    _Float16 lo = (_Float16)W[o * 128 + k];
    _Float16 hi = (_Float16)W[o * 128 + k + 1];
    uint32_t d = ((uint32_t)__builtin_bit_cast(uint16_t, hi) << 16) |
                 (uint32_t)__builtin_bit_cast(uint16_t, lo);
    F[idx] = d;
  }
}

// ---------------------------------------------------------------------------
// A-fragment chunk loaders. Each lane's 16 halves of a 16x32 f16 A tile are
// two contiguous 32B runs of the source f32 row: floats [k0..k0+7],
// k0 = hiK*2 + b*16  (hiK = 4 for lanes 16..31). Loaded as 2x b128 per run.
// ---------------------------------------------------------------------------
__device__ __forceinline__ void load_a_chunk(v16h& d, const float* __restrict__ base,
                                             int hiK) {
#pragma unroll
  for (int b = 0; b < 2; ++b) {
    int k0 = hiK * 2 + b * 16;
    v4f f0 = *(const v4f*)(base + k0);
    v4f f1 = *(const v4f*)(base + k0 + 4);
    d[8 * b + 0] = (_Float16)f0.x;  d[8 * b + 1] = (_Float16)f0.y;
    d[8 * b + 2] = (_Float16)f0.z;  d[8 * b + 3] = (_Float16)f0.w;
    d[8 * b + 4] = (_Float16)f1.x;  d[8 * b + 5] = (_Float16)f1.y;
    d[8 * b + 6] = (_Float16)f1.z;  d[8 * b + 7] = (_Float16)f1.w;
  }
}

__device__ __forceinline__ void load_a_chunk_nt(v16h& d, const float* __restrict__ base,
                                                int hiK) {
#pragma unroll
  for (int b = 0; b < 2; ++b) {
    int k0 = hiK * 2 + b * 16;
    v4f f0 = __builtin_nontemporal_load((const v4f*)(base + k0));
    v4f f1 = __builtin_nontemporal_load((const v4f*)(base + k0 + 4));
    d[8 * b + 0] = (_Float16)f0.x;  d[8 * b + 1] = (_Float16)f0.y;
    d[8 * b + 2] = (_Float16)f0.z;  d[8 * b + 3] = (_Float16)f0.w;
    d[8 * b + 4] = (_Float16)f1.x;  d[8 * b + 5] = (_Float16)f1.y;
    d[8 * b + 6] = (_Float16)f1.z;  d[8 * b + 7] = (_Float16)f1.w;
  }
}

// ---------------------------------------------------------------------------
// Shared GEMM core: 4 K-chunks x 4 N-tiles, c-outer / t-inner so the four
// WMMAs per c-step hit independent accumulators (pipelines the XDL unit);
// all 4 B fragments of a c-step are loaded from LDS before the WMMAs.
// ---------------------------------------------------------------------------
__device__ __forceinline__ void gemm_16x64(const v16h a[4],
                                           const uint32_t* sfrag, int lane,
                                           v8f acc[4]) {
#pragma unroll
  for (int c = 0; c < 4; ++c) {
    v16h bf0 = *(const v16h*)(sfrag + (((c * 4 + 0) * 32 + lane) << 3));
    v16h bf1 = *(const v16h*)(sfrag + (((c * 4 + 1) * 32 + lane) << 3));
    v16h bf2 = *(const v16h*)(sfrag + (((c * 4 + 2) * 32 + lane) << 3));
    v16h bf3 = *(const v16h*)(sfrag + (((c * 4 + 3) * 32 + lane) << 3));
    acc[0] = __builtin_amdgcn_wmma_f32_16x16x32_f16(false, a[c], false, bf0,
                                                    (short)0, acc[0], false, false);
    acc[1] = __builtin_amdgcn_wmma_f32_16x16x32_f16(false, a[c], false, bf1,
                                                    (short)0, acc[1], false, false);
    acc[2] = __builtin_amdgcn_wmma_f32_16x16x32_f16(false, a[c], false, bf2,
                                                    (short)0, acc[2], false, false);
    acc[3] = __builtin_amdgcn_wmma_f32_16x16x32_f16(false, a[c], false, bf3,
                                                    (short)0, acc[3], false, false);
  }
}

// ---------------------------------------------------------------------------
// Edge kernel: one wave = 16 edges x 64 outputs.
//   m = relu([nfeats[src] ‖ efeats] @ Wmsg^T + b); atomic scatter to
//   h_neigh[dst]. B fragments staged in LDS once per workgroup.
// ---------------------------------------------------------------------------
__global__ void __launch_bounds__(256)
edge_kernel(const float* __restrict__ nfeats,
            const float* __restrict__ efeats,
            const int*   __restrict__ src,
            const int*   __restrict__ dst,
            const uint32_t* __restrict__ frag,
            const float* __restrict__ bias,
            float* __restrict__ hneigh,
            int ntiles) {
  __shared__ uint32_t sfrag[4096];          // 16 KB B-fragment table
  {
    const uint4* gsrc = (const uint4*)frag;
    uint4*       sdst = (uint4*)sfrag;
#pragma unroll
    for (int i = 0; i < 4; ++i)
      sdst[threadIdx.x + 256 * i] = gsrc[threadIdx.x + 256 * i];
  }
  __syncthreads();                          // all waves participate

  int wid  = (int)((blockIdx.x * blockDim.x + threadIdx.x) >> 5);
  int lane = threadIdx.x & 31;
  if (wid >= ntiles) return;                // wave-uniform: EXEC stays all-ones

  int row = lane & 15;                      // A-matrix row owned by this lane
  int e   = wid * 16 + row;
  int sn  = src[e];
  const float* nrow = nfeats + (size_t)sn * DFEAT;
  const float* erow = efeats + (size_t)e * DFEAT;
  int hiK = (lane >= 16) ? 4 : 0;

  v16h a[4];
  load_a_chunk   (a[0], nrow,      hiK);    // K   0..31 (gathered node feats)
  load_a_chunk   (a[1], nrow + 32, hiK);    // K  32..63
  load_a_chunk_nt(a[2], erow,      hiK);    // K  64..95 (streamed edge feats)
  load_a_chunk_nt(a[3], erow + 32, hiK);    // K 96..127

  // Destination index base for each of the 8 C/D rows this lane owns
  int mbase = (lane >= 16) ? 8 : 0;
  uint32_t idx0[8];
#pragma unroll
  for (int v = 0; v < 8; ++v)
    idx0[v] = (uint32_t)dst[wid * 16 + mbase + v] * DFEAT + (uint32_t)row;

  float bv[4];
#pragma unroll
  for (int t = 0; t < 4; ++t) bv[t] = bias[t * 16 + row];

  v8f acc[4] = {v8f{}, v8f{}, v8f{}, v8f{}};
  gemm_16x64(a, sfrag, lane, acc);

#pragma unroll
  for (int t = 0; t < 4; ++t) {
#pragma unroll
    for (int v = 0; v < 8; ++v) {
      float val = fmaxf(acc[t][v] + bv[t], 0.0f);
      // base + 32-bit index; t*16 folds into the instruction offset
      unsafeAtomicAdd(hneigh + idx0[v] + t * 16, val);
    }
  }
}

// ---------------------------------------------------------------------------
// Node kernel: one wave = 16 nodes x 64 outputs.
//   out = relu([nfeats ‖ h_neigh] @ Wapply^T + b)
// ---------------------------------------------------------------------------
__global__ void __launch_bounds__(256)
node_kernel(const float* __restrict__ nfeats,
            const float* __restrict__ hneigh,
            const uint32_t* __restrict__ frag,
            const float* __restrict__ bias,
            float* __restrict__ out,
            int ntiles) {
  __shared__ uint32_t sfrag[4096];
  {
    const uint4* gsrc = (const uint4*)frag;
    uint4*       sdst = (uint4*)sfrag;
#pragma unroll
    for (int i = 0; i < 4; ++i)
      sdst[threadIdx.x + 256 * i] = gsrc[threadIdx.x + 256 * i];
  }
  __syncthreads();

  int wid  = (int)((blockIdx.x * blockDim.x + threadIdx.x) >> 5);
  int lane = threadIdx.x & 31;
  if (wid >= ntiles) return;                // wave-uniform

  int row  = lane & 15;
  int node = wid * 16 + row;
  const float* nrow = nfeats + (size_t)node * DFEAT;
  const float* hrow = hneigh + (size_t)node * DFEAT;
  int hiK = (lane >= 16) ? 4 : 0;

  v16h a[4];
  load_a_chunk(a[0], nrow,      hiK);       // K   0..31 (node feats)
  load_a_chunk(a[1], nrow + 32, hiK);       // K  32..63
  load_a_chunk(a[2], hrow,      hiK);       // K  64..95 (aggregated msgs)
  load_a_chunk(a[3], hrow + 32, hiK);       // K 96..127

  int mbase = (lane >= 16) ? 8 : 0;
  uint32_t obase[8];
#pragma unroll
  for (int v = 0; v < 8; ++v)
    obase[v] = (uint32_t)(wid * 16 + mbase + v) * DFEAT + (uint32_t)row;

  float bv[4];
#pragma unroll
  for (int t = 0; t < 4; ++t) bv[t] = bias[t * 16 + row];

  v8f acc[4] = {v8f{}, v8f{}, v8f{}, v8f{}};
  gemm_16x64(a, sfrag, lane, acc);

#pragma unroll
  for (int t = 0; t < 4; ++t) {
#pragma unroll
    for (int v = 0; v < 8; ++v) {
      float val = fmaxf(acc[t][v] + bv[t], 0.0f);
      // lanes 0..15 of a (t,v) pair write 64 contiguous bytes: coalesced
      __builtin_nontemporal_store(val, out + obase[v] + t * 16);
    }
  }
}

// ---------------------------------------------------------------------------
// Launch
// ---------------------------------------------------------------------------
extern "C" void kernel_launch(void* const* d_in, const int* in_sizes, int n_in,
                              void* d_out, int out_size, void* d_ws, size_t ws_size,
                              hipStream_t stream) {
  const float* nfeats = (const float*)d_in[0];
  const float* efeats = (const float*)d_in[1];
  const int*   src    = (const int*)d_in[2];
  const int*   dst    = (const int*)d_in[3];
  const float* Wm     = (const float*)d_in[4];
  const float* bm     = (const float*)d_in[5];
  const float* Wa     = (const float*)d_in[6];
  const float* ba     = (const float*)d_in[7];

  int nN = in_sizes[0] / DFEAT;    // 100000
  int nE = in_sizes[1] / DFEAT;    // 1600000

  // Workspace layout: [h_neigh: nN*64 f32][msg frags: 16KB][apply frags: 16KB]
  float*    hneigh = (float*)d_ws;
  size_t    hbytes = (size_t)nN * DFEAT * sizeof(float);
  uint32_t* fm     = (uint32_t*)((char*)d_ws + hbytes);
  uint32_t* fa     = fm + 4096;

  int n4 = nN * DFEAT / 4;
  zero_kernel<<<(n4 + 255) / 256, 256, 0, stream>>>((float4*)hneigh, n4);
  pack_weights<<<2, 256, 0, stream>>>(Wm, Wa, fm, fa);

  int etiles = nE / 16;            // 100000 tiles, divides exactly
  edge_kernel<<<(etiles * 32 + 255) / 256, 256, 0, stream>>>(
      nfeats, efeats, src, dst, fm, bm, hneigh, etiles);

  int vtiles = nN / 16;            // 6250 tiles
  node_kernel<<<(vtiles * 32 + 255) / 256, 256, 0, stream>>>(
      nfeats, hneigh, fa, ba, (float*)d_out, vtiles);
}
